// EnhancedGATConv_37168646979932
// MI455X (gfx1250) — matrix operand order repaired and checked by reading, not compiled, not run on hardware
//
#include <hip/hip_runtime.h>

// ---------------------------------------------------------------------------
// Only x2 is returned by the reference, so the CrossModal *edge* update
// branch is dead and skipped. Pipeline (stream-ordered, deterministic):
//   zero accumulators
//   node_q = x @ cm_n2e_q + b                (WMMA bf16, 16x64/wave)
//   edge_k = edge_attr @ cm_e2n_k + b
//   edge_v = edge_attr @ cm_e2n_v + b
//   CM seg-softmax (atomicMax -> expsum -> normalized scatter)
//   x1 = agg @ cm_node_out + b + x
//   q,k,v = x1 @ rp_{q,k,v} + b
//   RP seg-softmax with distance-binned positional tables
//   x2 = outn @ rp_out + b + x1 -> d_out
// ---------------------------------------------------------------------------

#define NN 20000
#define NE 320000
#define MAXD 100

typedef __attribute__((ext_vector_type(16))) __bf16 v16bf;
typedef __attribute__((ext_vector_type(8)))  float  v8f;

// ---- WMMA GEMM: out[M,Nout] = A[M,K] @ W[K,Nout] + bias (+resid) ----------
// One wave per block; each wave owns a 16x64 output tile (4 accumulators).
// LDS tiles are stored in *fragment order* so each lane's WMMA operand is a
// single contiguous 32-byte LDS read (2x ds_load_b128), no scalar gathers.
__global__ __launch_bounds__(32) void gemm_wmma_bf16_kernel(
    const float* __restrict__ A, const float* __restrict__ W,
    const float* __restrict__ bias, const float* __restrict__ resid,
    float* __restrict__ out, int M, int K, int Nout)
{
    __shared__ alignas(32) __bf16 lA[16 * 32];        // one 16x32 A tile
    __shared__ alignas(32) __bf16 lB[4 * 16 * 32];    // four 32x16 B tiles
    const int lane  = threadIdx.x;                    // wave32
    const int mtile = blockIdx.x * 16;
    const int ncol0 = blockIdx.y * 64;
    const int mm = lane & 15;
    const int hi = lane >> 4;

    // A-store mapping for element (row r, k c): frag lane/elem (ISA 7.12.2)
    const int aDL = ((lane >> 3) & 1) * 16;           // + r
    const int aDJ = (lane & 7) | ((lane >> 4) << 3);

    v8f acc0 = {}, acc1 = {}, acc2 = {}, acc3 = {};
    for (int k0 = 0; k0 < K; k0 += 32) {
        // ---- stage A tile (r=t, c=lane; fully coalesced) ----
#pragma unroll
        for (int t = 0; t < 16; ++t) {
            lA[(t + aDL) * 16 + aDJ] = (__bf16)A[(mtile + t) * K + k0 + lane];
        }
        // ---- stage W tile 32x64 ----
#pragma unroll 4
        for (int r = 0; r < 32; ++r) {
            const int base = (k0 + r) * Nout + ncol0;
            const int dlk  = ((r >> 3) & 1) * 16;
            const int dj   = (r & 7) | ((r >> 4) << 3);
#pragma unroll
            for (int ha = 0; ha < 2; ++ha) {
                int c  = ha * 32 + lane;
                int nt = c >> 4, nn = c & 15;
                lB[nt * 512 + (nn + dlk) * 16 + dj] = (__bf16)W[base + c];
            }
        }
        __syncthreads();
        v16bf af = *(const v16bf*)&lA[lane * 16];
        v16bf b0 = *(const v16bf*)&lB[0 * 512 + lane * 16];
        v16bf b1 = *(const v16bf*)&lB[1 * 512 + lane * 16];
        v16bf b2 = *(const v16bf*)&lB[2 * 512 + lane * 16];
        v16bf b3 = *(const v16bf*)&lB[3 * 512 + lane * 16];
        acc0 = __builtin_amdgcn_wmma_f32_16x16x32_bf16(false, af, false, b0, (short)0, acc0, false, false);
        acc1 = __builtin_amdgcn_wmma_f32_16x16x32_bf16(false, af, false, b1, (short)0, acc1, false, false);
        acc2 = __builtin_amdgcn_wmma_f32_16x16x32_bf16(false, af, false, b2, (short)0, acc2, false, false);
        acc3 = __builtin_amdgcn_wmma_f32_16x16x32_bf16(false, af, false, b3, (short)0, acc3, false, false);
        __syncthreads();
    }
    // ---- epilogue: D layout VGPR r -> M=r+8*hi, N=mm ----
#pragma unroll
    for (int t = 0; t < 4; ++t) {
        v8f a = (t == 0) ? acc0 : (t == 1) ? acc1 : (t == 2) ? acc2 : acc3;
        int n = ncol0 + t * 16 + mm;
        float bn = bias[n];
#pragma unroll
        for (int r = 0; r < 8; ++r) {
            int m = mtile + r + (hi << 3);
            float val = a[r] + bn;
            if (resid) val += resid[m * Nout + n];
            out[m * Nout + n] = val;
        }
    }
}

// ---------------- helpers --------------------------------------------------
__global__ void fill_zero_kernel(float* __restrict__ p, size_t n) {
    size_t i = blockIdx.x * (size_t)blockDim.x + threadIdx.x;
    if (i < n) p[i] = 0.0f;
}

__device__ __forceinline__ unsigned ordEnc(float f) {
    unsigned u = __float_as_uint(f);
    return (u & 0x80000000u) ? ~u : (u | 0x80000000u);
}
__device__ __forceinline__ float ordDec(unsigned u) {
    return __uint_as_float((u & 0x80000000u) ? (u ^ 0x80000000u) : ~u);
}

__device__ __forceinline__ float dot16(const float4* __restrict__ a,
                                       const float4* __restrict__ b) {
    float s = 0.f;
#pragma unroll
    for (int i = 0; i < 4; ++i) {
        float4 x = a[i], y = b[i];
        s += x.x * y.x + x.y * y.y + x.z * y.z + x.w * y.w;
    }
    return s;
}

// ---------------- CrossModal node update (4 heads x 16 dims) ---------------
__global__ void cm_score_kernel(const float* __restrict__ nq,
                                const float* __restrict__ ek,
                                const int* __restrict__ row,
                                const int* __restrict__ col,
                                float* __restrict__ s2,
                                unsigned* __restrict__ cmmax)
{
    long long gid = blockIdx.x * (long long)blockDim.x + threadIdx.x;
    if (gid >= 2LL * NE * 4) return;
    int h = (int)(gid & 3);
    long long e2 = gid >> 2;
    int e    = (e2 < NE) ? (int)e2 : (int)(e2 - NE);
    int node = (e2 < NE) ? row[e] : col[e];
    const float4* qp = (const float4*)(nq + node * 64 + h * 16);
    const float4* kp = (const float4*)(ek + (size_t)e * 64 + h * 16);
    __builtin_prefetch(kp + 16, 0, 1);   // global_prefetch_b8
    float s = dot16(qp, kp) * 0.25f;     // Dc^-0.5, Dc=16
    s2[e2 * 4 + h] = s;
    atomicMax(&cmmax[node * 4 + h], ordEnc(s));
}

__global__ void cm_expsum_kernel(const float* __restrict__ s2,
                                 const int* __restrict__ row,
                                 const int* __restrict__ col,
                                 const unsigned* __restrict__ cmmax,
                                 float* __restrict__ cmz)
{
    long long gid = blockIdx.x * (long long)blockDim.x + threadIdx.x;
    if (gid >= 2LL * NE * 4) return;
    int h = (int)(gid & 3);
    long long e2 = gid >> 2;
    int e    = (e2 < NE) ? (int)e2 : (int)(e2 - NE);
    int node = (e2 < NE) ? row[e] : col[e];
    float w = __expf(s2[e2 * 4 + h] - ordDec(cmmax[node * 4 + h]));
    atomicAdd(&cmz[node * 4 + h], w);
}

__global__ void cm_accum_kernel(const float* __restrict__ s2,
                                const float* __restrict__ ev,
                                const int* __restrict__ row,
                                const int* __restrict__ col,
                                const unsigned* __restrict__ cmmax,
                                const float* __restrict__ cmz,
                                float* __restrict__ agg)
{
    long long gid = blockIdx.x * (long long)blockDim.x + threadIdx.x;
    if (gid >= 2LL * NE * 4) return;
    int h = (int)(gid & 3);
    long long e2 = gid >> 2;
    int e    = (e2 < NE) ? (int)e2 : (int)(e2 - NE);
    int node = (e2 < NE) ? row[e] : col[e];
    float w = __expf(s2[e2 * 4 + h] - ordDec(cmmax[node * 4 + h])) /
              (cmz[node * 4 + h] + 1e-16f);
    const float4* vp = (const float4*)(ev + (size_t)e * 64 + h * 16);
    float* ap = agg + node * 64 + h * 16;
#pragma unroll
    for (int i = 0; i < 4; ++i) {
        float4 vv = vp[i];
        atomicAdd(&ap[i * 4 + 0], w * vv.x);
        atomicAdd(&ap[i * 4 + 1], w * vv.y);
        atomicAdd(&ap[i * 4 + 2], w * vv.z);
        atomicAdd(&ap[i * 4 + 3], w * vv.w);
    }
}

// ---------------- RelativePositionalAttention (8 heads x 16 dims) ----------
__global__ void rp_score_kernel(const float* __restrict__ q,
                                const float* __restrict__ k,
                                const float* __restrict__ pos,
                                const float* __restrict__ pos_k,
                                const int* __restrict__ row,
                                const int* __restrict__ col,
                                float* __restrict__ sc,
                                int* __restrict__ bins,
                                unsigned* __restrict__ rpmax)
{
    long long gid = blockIdx.x * (long long)blockDim.x + threadIdx.x;
    if (gid >= (long long)NE * 8) return;
    int h = (int)(gid & 7);
    int e = (int)(gid >> 3);
    int r = row[e], c = col[e];
    float dx = pos[r * 3 + 0] - pos[c * 3 + 0];
    float dy = pos[r * 3 + 1] - pos[c * 3 + 1];
    float dz = pos[r * 3 + 2] - pos[c * 3 + 2];
    float dist = sqrtf(dx * dx + dy * dy + dz * dz);
    int b = (int)(dist * 10.0f);                 // trunc == astype(int32)
    b = (b > MAXD) ? MAXD : ((b < -MAXD) ? -MAXD : b);
    b += MAXD;
    if (h == 0) bins[e] = b;
    const float4* qp = (const float4*)(q + r * 128 + h * 16);
    const float4* kp = (const float4*)(k + c * 128 + h * 16);
    const float4* rk = (const float4*)(pos_k + b * 128 + h * 16);
    __builtin_prefetch(qp + 32, 0, 1);
    float s = 0.f;
#pragma unroll
    for (int i = 0; i < 4; ++i) {
        float4 qa = qp[i], kb = kp[i], rb = rk[i];
        s += qa.x * (kb.x + rb.x) + qa.y * (kb.y + rb.y)
           + qa.z * (kb.z + rb.z) + qa.w * (kb.w + rb.w);
    }
    s *= 0.25f;                                  // Dr^-0.5, Dr=16
    sc[(size_t)e * 8 + h] = s;
    atomicMax(&rpmax[r * 8 + h], ordEnc(s));
}

__global__ void rp_expsum_kernel(const float* __restrict__ sc,
                                 const int* __restrict__ row,
                                 const unsigned* __restrict__ rpmax,
                                 float* __restrict__ rpz)
{
    long long gid = blockIdx.x * (long long)blockDim.x + threadIdx.x;
    if (gid >= (long long)NE * 8) return;
    int h = (int)(gid & 7);
    int e = (int)(gid >> 3);
    int r = row[e];
    float w = __expf(sc[(size_t)e * 8 + h] - ordDec(rpmax[r * 8 + h]));
    atomicAdd(&rpz[r * 8 + h], w);
}

__global__ void rp_accum_kernel(const float* __restrict__ sc,
                                const float* __restrict__ v,
                                const float* __restrict__ pos_v,
                                const int* __restrict__ row,
                                const int* __restrict__ col,
                                const int* __restrict__ bins,
                                const unsigned* __restrict__ rpmax,
                                const float* __restrict__ rpz,
                                float* __restrict__ outn)
{
    long long gid = blockIdx.x * (long long)blockDim.x + threadIdx.x;
    if (gid >= (long long)NE * 8) return;
    int h = (int)(gid & 7);
    int e = (int)(gid >> 3);
    int r = row[e], c = col[e], b = bins[e];
    float w = __expf(sc[(size_t)e * 8 + h] - ordDec(rpmax[r * 8 + h])) /
              (rpz[r * 8 + h] + 1e-16f);
    const float4* vp = (const float4*)(v + c * 128 + h * 16);
    const float4* rv = (const float4*)(pos_v + b * 128 + h * 16);
    float* op = outn + r * 128 + h * 16;
#pragma unroll
    for (int i = 0; i < 4; ++i) {
        float4 vv = vp[i], rr = rv[i];
        atomicAdd(&op[i * 4 + 0], w * (vv.x + rr.x));
        atomicAdd(&op[i * 4 + 1], w * (vv.y + rr.y));
        atomicAdd(&op[i * 4 + 2], w * (vv.z + rr.z));
        atomicAdd(&op[i * 4 + 3], w * (vv.w + rr.w));
    }
}

// ---------------- driver ---------------------------------------------------
extern "C" void kernel_launch(void* const* d_in, const int* in_sizes, int n_in,
                              void* d_out, int out_size, void* d_ws, size_t ws_size,
                              hipStream_t stream)
{
    (void)in_sizes; (void)n_in; (void)out_size; (void)ws_size;
    const float* x         = (const float*)d_in[0];
    const float* edge_attr = (const float*)d_in[1];
    const int*   eidx      = (const int*)d_in[2];
    const int*   row       = eidx;
    const int*   col       = eidx + NE;
    const float* node_pos  = (const float*)d_in[3];
    const float* cmq_w  = (const float*)d_in[4];  const float* cmq_b  = (const float*)d_in[5];
    const float* cek_w  = (const float*)d_in[12]; const float* cek_b  = (const float*)d_in[13];
    const float* cev_w  = (const float*)d_in[14]; const float* cev_b  = (const float*)d_in[15];
    const float* cno_w  = (const float*)d_in[16]; const float* cno_b  = (const float*)d_in[17];
    const float* rpq_w  = (const float*)d_in[20]; const float* rpq_b  = (const float*)d_in[21];
    const float* rpk_w  = (const float*)d_in[22]; const float* rpk_b  = (const float*)d_in[23];
    const float* rpv_w  = (const float*)d_in[24]; const float* rpv_b  = (const float*)d_in[25];
    const float* rpo_w  = (const float*)d_in[26]; const float* rpo_b  = (const float*)d_in[27];
    const float* pos_k  = (const float*)d_in[28];
    const float* pos_v  = (const float*)d_in[29];

    float* ws = (float*)d_ws;
    size_t o = 0;
    float* node_q = ws + o; o += (size_t)NN * 64;
    float* edge_k = ws + o; o += (size_t)NE * 64;
    float* edge_v = ws + o; o += (size_t)NE * 64;
    float* s2     = ws + o; o += (size_t)2 * NE * 4;
    float* sc     = ws + o; o += (size_t)NE * 8;
    int*   bins   = (int*)(ws + o); o += (size_t)NE;
    float* x1     = ws + o; o += (size_t)NN * 128;
    float* q      = ws + o; o += (size_t)NN * 128;
    float* k      = ws + o; o += (size_t)NN * 128;
    float* v      = ws + o; o += (size_t)NN * 128;
    float* zreg   = ws + o;                    // contiguous zero-init region
    unsigned* cm_max = (unsigned*)(ws + o); o += (size_t)NN * 4;
    float*    cm_z   = ws + o;              o += (size_t)NN * 4;
    float*    agg    = ws + o;              o += (size_t)NN * 64;
    unsigned* rp_max = (unsigned*)(ws + o); o += (size_t)NN * 8;
    float*    rp_z   = ws + o;              o += (size_t)NN * 8;
    float*    outn   = ws + o;              o += (size_t)NN * 128;
    size_t zcount = (size_t)NN * (4 + 4 + 64 + 8 + 8 + 128);

    fill_zero_kernel<<<(unsigned)((zcount + 255) / 256), 256, 0, stream>>>(zreg, zcount);

    // ---- projections (WMMA, 16x64 tile per wave) ----
    gemm_wmma_bf16_kernel<<<dim3(NN / 16, 1), 32, 0, stream>>>(
        x, cmq_w, cmq_b, nullptr, node_q, NN, 128, 64);
    gemm_wmma_bf16_kernel<<<dim3(NE / 16, 1), 32, 0, stream>>>(
        edge_attr, cek_w, cek_b, nullptr, edge_k, NE, 64, 64);
    gemm_wmma_bf16_kernel<<<dim3(NE / 16, 1), 32, 0, stream>>>(
        edge_attr, cev_w, cev_b, nullptr, edge_v, NE, 64, 64);

    // ---- CM segmented softmax over (edge,endpoint) pairs ----
    long long cmT = 2LL * NE * 4;
    unsigned cmB = (unsigned)((cmT + 255) / 256);
    cm_score_kernel <<<cmB, 256, 0, stream>>>(node_q, edge_k, row, col, s2, cm_max);
    cm_expsum_kernel<<<cmB, 256, 0, stream>>>(s2, row, col, cm_max, cm_z);
    cm_accum_kernel <<<cmB, 256, 0, stream>>>(s2, edge_v, row, col, cm_max, cm_z, agg);

    gemm_wmma_bf16_kernel<<<dim3(NN / 16, 2), 32, 0, stream>>>(
        agg, cno_w, cno_b, x, x1, NN, 64, 128);

    // ---- RP projections (WMMA) ----
    gemm_wmma_bf16_kernel<<<dim3(NN / 16, 2), 32, 0, stream>>>(
        x1, rpq_w, rpq_b, nullptr, q, NN, 128, 128);
    gemm_wmma_bf16_kernel<<<dim3(NN / 16, 2), 32, 0, stream>>>(
        x1, rpk_w, rpk_b, nullptr, k, NN, 128, 128);
    gemm_wmma_bf16_kernel<<<dim3(NN / 16, 2), 32, 0, stream>>>(
        x1, rpv_w, rpv_b, nullptr, v, NN, 128, 128);

    // ---- RP segmented softmax over edges (seg = row) ----
    long long rpT = (long long)NE * 8;
    unsigned rpB = (unsigned)((rpT + 255) / 256);
    rp_score_kernel <<<rpB, 256, 0, stream>>>(q, k, node_pos, pos_k, row, col,
                                              sc, bins, rp_max);
    rp_expsum_kernel<<<rpB, 256, 0, stream>>>(sc, row, rp_max, rp_z);
    rp_accum_kernel <<<rpB, 256, 0, stream>>>(sc, v, pos_v, row, col, bins,
                                              rp_max, rp_z, outn);

    // ---- final projection + residual -> d_out ----
    gemm_wmma_bf16_kernel<<<dim3(NN / 16, 2), 32, 0, stream>>>(
        outn, rpo_w, rpo_b, x1, (float*)d_out, NN, 128, 128);
}
